// SSA_70076686402468
// MI455X (gfx1250) — matrix operand order repaired and checked by reading, not compiled, not run on hardware
//
#include <hip/hip_runtime.h>
#include <hip/hip_bf16.h>
#include <math.h>

// ---------------------------------------------------------------------------
// PVT spatial-reduction attention for MI455X (gfx1250), wave32 + WMMA f16.
// B=16, N=3136(56x56), C=256, HEADS=8, SR=2, Nkv=784, C_NEW=320, dq=40, dv=32
// ---------------------------------------------------------------------------

typedef _Float16 half_t;
typedef __attribute__((ext_vector_type(16))) _Float16 v16h;
typedef __attribute__((ext_vector_type(8)))  _Float16 v8h;
typedef __attribute__((ext_vector_type(8)))  float    v8f;

#define WMMA_F16(A, Bm, Cm) \
  __builtin_amdgcn_wmma_f32_16x16x32_f16(false, (A), false, (Bm), (short)0, (Cm), false, false)

static __device__ __forceinline__ v16h cat8(v8h lo, v8h hi) {
  return __builtin_shufflevector(lo, hi, 0,1,2,3,4,5,6,7,8,9,10,11,12,13,14,15);
}

// ---------------------------------------------------------------------------
// Weight transpose + f32->f16 convert: WT[j*in_dim + c] = W[c*out_dim + j]*scale
// ---------------------------------------------------------------------------
__global__ void wtrans_kernel(const float* __restrict__ W, half_t* __restrict__ WT,
                              int in_dim, int out_dim, float scale) {
  int idx = blockIdx.x * 256 + threadIdx.x;
  if (idx >= in_dim * out_dim) return;
  int j = idx / in_dim, c = idx - j * in_dim;
  WT[idx] = (half_t)(W[c * out_dim + j] * scale);
}

__global__ void f2h_kernel(const float* __restrict__ src, half_t* __restrict__ dst, int n) {
  int i = blockIdx.x * 256 + threadIdx.x;
  if (i < n) dst[i] = (half_t)src[i];
}

// ---------------------------------------------------------------------------
// Spatial reduction: depthwise 2x2 stride-2 conv + 1x1 conv(C->320) + LN + GELU
// grid = B*784 blocks, 320 threads. Output xr: [B, 784, 320] f16.
// ---------------------------------------------------------------------------
__global__ void sr_kernel(const float* __restrict__ x, const float* __restrict__ dwk,
                          const float* __restrict__ dwb, const float* __restrict__ pwk,
                          const float* __restrict__ pwb, const float* __restrict__ lng,
                          const float* __restrict__ lnb, half_t* __restrict__ xr) {
  __shared__ float y[256];
  __shared__ float s1[320], s2[320];
  const int b = blockIdx.x / 784;
  const int p = blockIdx.x - b * 784;
  const int oh = p / 28, ow = p - oh * 28;
  const int t = threadIdx.x;
  if (t < 256) {
    const int c = t;
    float acc = dwb[c];
#pragma unroll
    for (int i = 0; i < 2; ++i)
#pragma unroll
      for (int jx = 0; jx < 2; ++jx) {
        const int n = (2 * oh + i) * 56 + 2 * ow + jx;
        acc += x[(((size_t)b * 3136) + n) * 256 + c] * dwk[c * 4 + i * 2 + jx];
      }
    y[c] = acc;
  }
  __syncthreads();
  // pointwise 1x1 conv: thread t computes output channel j=t (0..319)
  float z = pwb[t];
  for (int c = 0; c < 256; ++c) z += y[c] * pwk[t * 256 + c];
  s1[t] = z; s2[t] = z * z;
  __syncthreads();
  if (t < 64) { s1[t] += s1[t + 256]; s2[t] += s2[t + 256]; }
  __syncthreads();
  for (int st = 128; st > 0; st >>= 1) {
    if (t < st) { s1[t] += s1[t + st]; s2[t] += s2[t + st]; }
    __syncthreads();
  }
  const float mu  = s1[0] * (1.f / 320.f);
  const float var = s2[0] * (1.f / 320.f) - mu * mu;
  const float zn  = (z - mu) * rsqrtf(var + 1e-5f) * lng[t] + lnb[t];
  const float g   = 0.5f * zn * (1.f + erff(zn * 0.70710678118654752f));
  xr[(((size_t)b * 784) + p) * 320 + t] = (half_t)g;
}

// ---------------------------------------------------------------------------
// Generic 16x16-tile WMMA GEMM, one wave per tile. A:[rows,lda] f16 row-major,
// BT:[cols,lda] f16 (transposed weights, so B-layout loads are contiguous).
// mode 0: store k  [B,H,800,40]   (cols 320 -> h=j/40, d=j%40, rows = b*784+p)
// mode 1: store vT [B,H,32,800]   (cols 256 -> h=j/32, dv=j%32)
// mode 2: store q  [B,H,3136,40]  (cols 320, rows = b*3136+n)
// mode 3: store f32 out[row*256+j] + bias[j]
// ---------------------------------------------------------------------------
__global__ void gemm16_kernel(const half_t* __restrict__ A, int lda,
                              const half_t* __restrict__ BT, int mode,
                              const float* __restrict__ bias,
                              half_t* __restrict__ outh, float* __restrict__ outf) {
  const int lane = threadIdx.x;
  const int m = lane & 15, hi = lane >> 4;
  const int row0 = blockIdx.x * 16, j0 = blockIdx.y * 16;
  const half_t* arow = A + (size_t)(row0 + m) * lda;
  const half_t* brow = BT + (size_t)(j0 + m) * lda;
  v8f acc = {};
  const int nk = lda >> 5;
  for (int s = 0; s < nk; ++s) {
    const int cb = s * 32;
    // A layout (16-bit 16x32): lane element e -> K = (e<8?e:e+8) + 8*hi
    v8h alo = *(const v8h*)(arow + cb + hi * 8);
    v8h ahi = *(const v8h*)(arow + cb + 16 + hi * 8);
    // B layout (16-bit 32x16): lane element e -> K = (e<8?e:e+8) + 8*hi, col = m
    v8h blo = *(const v8h*)(brow + cb + hi * 8);
    v8h bhi = *(const v8h*)(brow + cb + 16 + hi * 8);
    acc = WMMA_F16(cat8(alo, ahi), cat8(blo, bhi), acc);
  }
  const int j = j0 + m;
#pragma unroll
  for (int r = 0; r < 8; ++r) {
    const int row = row0 + r + hi * 8;   // C layout: slot r -> row r + 8*hi, col m
    const float val = acc[r];
    if (mode == 0) {
      int b = row / 784, p = row - b * 784;
      int h = j / 40, d = j - h * 40;
      outh[(((size_t)(b * 8 + h) * 800) + p) * 40 + d] = (half_t)val;
    } else if (mode == 1) {
      int b = row / 784, p = row - b * 784;
      int h = j / 32, dv = j - h * 32;
      outh[(((size_t)(b * 8 + h) * 32) + dv) * 800 + p] = (half_t)val;
    } else if (mode == 2) {
      int b = row / 3136, n = row - b * 3136;
      int h = j / 40, d = j - h * 40;
      outh[(((size_t)(b * 8 + h) * 3136) + n) * 40 + d] = (half_t)val;
    } else {
      outf[(size_t)row * 256 + j] = val + bias[j];
    }
  }
}

// ---------------------------------------------------------------------------
// Flash attention, one wave per (b, h, 16-query-row tile). 25 kv-chunks of 32
// (kv padded to 800 with zeroed k/v; padded columns masked to -1e30).
// q scale folded in. dq=40 padded to 64 via register zeros.
// ---------------------------------------------------------------------------
__global__ void attn_kernel(const half_t* __restrict__ q, const half_t* __restrict__ k,
                            const half_t* __restrict__ vT, half_t* __restrict__ attnout) {
  __shared__ __align__(16) half_t pst[16 * 32];   // P tile staging (C->A transpose)
  const int bid = blockIdx.x;
  const int qt = bid % 196;
  const int h  = (bid / 196) & 7;
  const int b  = bid / 1568;
  const int lane = threadIdx.x;
  const int m = lane & 15, hi = lane >> 4;

  const half_t* qrow  = q  + (((size_t)(b * 8 + h)) * 3136 + qt * 16 + m) * 40;
  const half_t* kbase = k  + ((size_t)(b * 8 + h)) * 800 * 40;
  const half_t* vbase = vT + ((size_t)(b * 8 + h)) * 32 * 800;

  // Load q once in A layout; k-step1 covers dq 32..63 (valid halves only).
  v16h qa0 = cat8(*(const v8h*)(qrow + hi * 8), *(const v8h*)(qrow + 16 + hi * 8));
  v16h qa1 = {};
  if (hi == 0) {
    v8h qtail = *(const v8h*)(qrow + 32);
#pragma unroll
    for (int e = 0; e < 8; ++e) qa1[e] = qtail[e];
  }

  float mst[8], lst[8];
#pragma unroll
  for (int r = 0; r < 8; ++r) { mst[r] = -1e30f; lst[r] = 0.f; }
  v8f o0 = {}, o1 = {};

  for (int ch = 0; ch < 25; ++ch) {
    const int kvb = ch * 32;
    v8f s0, s1v;
#pragma unroll
    for (int t = 0; t < 2; ++t) {
      const half_t* krow = kbase + (size_t)(kvb + t * 16 + m) * 40;
      v16h b0 = cat8(*(const v8h*)(krow + hi * 8), *(const v8h*)(krow + 16 + hi * 8));
      v16h b1 = {};
      if (hi == 0) {
        v8h ktail = *(const v8h*)(krow + 32);
#pragma unroll
        for (int e = 0; e < 8; ++e) b1[e] = ktail[e];
      }
      v8f st = {};
      st = WMMA_F16(qa0, b0, st);
      st = WMMA_F16(qa1, b1, st);
      if (kvb + t * 16 + m >= 784) {         // mask padded kv columns
#pragma unroll
        for (int r = 0; r < 8; ++r) st[r] = -1e30f;
      }
      if (t == 0) s0 = st; else s1v = st;
    }
    // Online softmax: row r+8*hi lives in slot r across 16 lanes of this half.
#pragma unroll
    for (int r = 0; r < 8; ++r) {
      float cm = fmaxf(s0[r], s1v[r]);
      for (int off = 8; off >= 1; off >>= 1)
        cm = fmaxf(cm, __shfl_xor(cm, off, 32));
      const float nm = fmaxf(mst[r], cm);
      const float alpha = __expf(mst[r] - nm);
      mst[r] = nm;
      const float p0 = __expf(s0[r] - nm);
      const float p1 = __expf(s1v[r] - nm);
      float ps = p0 + p1;
      for (int off = 8; off >= 1; off >>= 1)
        ps += __shfl_xor(ps, off, 32);
      lst[r] = lst[r] * alpha + ps;
      o0[r] *= alpha;
      o1[r] *= alpha;
      pst[(r + hi * 8) * 32 + m]      = (half_t)p0;
      pst[(r + hi * 8) * 32 + 16 + m] = (half_t)p1;
    }
    __syncthreads();
    // Reload P in A layout: element e -> kv = (e<8?e:e+8) + 8*hi
    v16h pa = cat8(*(const v8h*)(pst + m * 32 + hi * 8),
                   *(const v8h*)(pst + m * 32 + 16 + hi * 8));
#pragma unroll
    for (int t = 0; t < 2; ++t) {
      // V in B layout from vT: element e -> kv = (e<8?e:e+8) + 8*hi, col dv = t*16+m
      const half_t* vrow = vbase + (size_t)(t * 16 + m) * 800 + kvb + hi * 8;
      v16h bv = cat8(*(const v8h*)(vrow), *(const v8h*)(vrow + 16));
      if (t == 0) o0 = WMMA_F16(pa, bv, o0);
      else        o1 = WMMA_F16(pa, bv, o1);
    }
    __syncthreads();
  }
#pragma unroll
  for (int r = 0; r < 8; ++r) {
    const float inv = 1.f / lst[r];
    const int n = qt * 16 + r + hi * 8;
    const size_t base = ((size_t)b * 3136 + n) * 256 + h * 32;
    attnout[base + m]      = (half_t)(o0[r] * inv);
    attnout[base + 16 + m] = (half_t)(o1[r] * inv);
  }
}

// ---------------------------------------------------------------------------
extern "C" void kernel_launch(void* const* d_in, const int* in_sizes, int n_in,
                              void* d_out, int out_size, void* d_ws, size_t ws_size,
                              hipStream_t stream) {
  const float* x     = (const float*)d_in[0];
  const float* dwk   = (const float*)d_in[1];
  const float* dwb   = (const float*)d_in[2];
  const float* pwk   = (const float*)d_in[3];
  const float* pwb   = (const float*)d_in[4];
  const float* lng   = (const float*)d_in[5];
  const float* lnb   = (const float*)d_in[6];
  const float* Wq    = (const float*)d_in[7];
  const float* Wk    = (const float*)d_in[8];
  const float* Wv    = (const float*)d_in[9];
  const float* Wproj = (const float*)d_in[10];
  const float* bproj = (const float*)d_in[11];
  float* out = (float*)d_out;

  char* ws = (char*)d_ws;
  size_t off = 0;
  auto alloc = [&](size_t bytes) -> void* {
    void* p = ws + off;
    off += (bytes + 255) & ~(size_t)255;
    return p;
  };
  half_t* xh  = (half_t*)alloc(50176ull * 256 * 2);      // x f16
  half_t* xr  = (half_t*)alloc(16ull * 784 * 320 * 2);   // spatial-reduced f16
  half_t* qb  = (half_t*)alloc(16ull * 8 * 3136 * 40 * 2);
  half_t* kb  = (half_t*)alloc(16ull * 8 * 800 * 40 * 2);
  half_t* vb  = (half_t*)alloc(16ull * 8 * 32 * 800 * 2);
  half_t* ao  = (half_t*)alloc(50176ull * 256 * 2);      // attention out f16
  half_t* WqT = (half_t*)alloc(320ull * 256 * 2);
  half_t* WkT = (half_t*)alloc(320ull * 320 * 2);
  half_t* WvT = (half_t*)alloc(256ull * 320 * 2);
  half_t* WpT = (half_t*)alloc(256ull * 256 * 2);

  const float SCALE = 0.15811388300841897f;  // (40)^-0.5, folded into Wq

  wtrans_kernel<<<(320 * 256 + 255) / 256, 256, 0, stream>>>(Wq, WqT, 256, 320, SCALE);
  wtrans_kernel<<<(320 * 320 + 255) / 256, 256, 0, stream>>>(Wk, WkT, 320, 320, 1.f);
  wtrans_kernel<<<(256 * 320 + 255) / 256, 256, 0, stream>>>(Wv, WvT, 320, 256, 1.f);
  wtrans_kernel<<<(256 * 256 + 255) / 256, 256, 0, stream>>>(Wproj, WpT, 256, 256, 1.f);
  f2h_kernel<<<(12845056 + 255) / 256, 256, 0, stream>>>(x, xh, 12845056);
  sr_kernel<<<16 * 784, 320, 0, stream>>>(x, dwk, dwb, pwk, pwb, lng, lnb, xr);

  hipMemsetAsync(kb, 0, 16ull * 8 * 800 * 40 * 2, stream);  // kv rows 784..799 = 0
  hipMemsetAsync(vb, 0, 16ull * 8 * 32 * 800 * 2, stream);

  gemm16_kernel<<<dim3(3136, 20), 32, 0, stream>>>(xh, 256, WqT, 2, nullptr, qb, nullptr);
  gemm16_kernel<<<dim3(784, 20),  32, 0, stream>>>(xr, 320, WkT, 0, nullptr, kb, nullptr);
  gemm16_kernel<<<dim3(784, 16),  32, 0, stream>>>(xr, 320, WvT, 1, nullptr, vb, nullptr);

  attn_kernel<<<16 * 8 * 196, 32, 0, stream>>>(qb, kb, vb, ao);

  gemm16_kernel<<<dim3(3136, 16), 32, 0, stream>>>(ao, 256, WpT, 3, bproj, nullptr, out);
}
